// LogVAE_47682726920677
// MI455X (gfx1250) — matrix operand order repaired and checked by reading, not compiled, not run on hardware
//
#include <hip/hip_runtime.h>
#include <stdint.h>

// ---------------------------------------------------------------------------
// Problem constants
// ---------------------------------------------------------------------------
#define B_   256
#define S_   1024
#define NF_  32
#define E_   64
#define H_   128
#define L_   64
#define C_   31
#define G_   512          // 4*H
#define SD   1023         // S-1 (decoder steps)

// d_out layout: acts | ts | mu | logvar  (all f32)
#define ACTS_CNT  (B_ * SD * C_)          // 8,118,528
#define TS_OFF    (ACTS_CNT)              // 8,118,528
#define MU_OFF    (TS_OFF + B_ * SD)      // 8,380,416
#define LV_OFF    (MU_OFF + B_ * L_)      // 8,396,800

typedef __attribute__((ext_vector_type(16))) _Float16 v16h;
typedef __attribute__((ext_vector_type(8)))  float    v8f;

union Frag16 {
    v16h     h;
    uint32_t u[8];
    uint4    q[2];
};

// A-fragment K offset for 16-bit 16x32 A matrix (ISA 7.12.2):
// lanes 0-15 : V0..V3 = K{0..7},  V4..V7 = K{16..23}
// lanes 16-31: V0..V3 = K{8..15}, V4..V7 = K{24..31}
static __device__ __forceinline__ int a_koff(int j, int hi) {
    return ((j & 3) << 1) + (hi << 3) + ((j >> 2) << 4);
}

// Fast sigmoid/tanh: v_exp_f32 + v_rcp_f32 (both TRANS32, co-issue with VALU),
// avoids the IEEE v_div_scale/div_fmas expansion.
static __device__ __forceinline__ float fsig(float x) {
    return __builtin_amdgcn_rcpf(1.0f + __expf(-x));
}
static __device__ __forceinline__ float ftanh(float x) {
    x = fminf(fmaxf(x, -15.0f), 15.0f);
    float e = __expf(2.0f * x);
    return (e - 1.0f) * __builtin_amdgcn_rcpf(e + 1.0f);
}

// ---------------------------------------------------------------------------
// Kernel 1: pack weights into WMMA B-fragment layout + fold biases
//   B-fragment (16-bit 32x16): lane n = l&15, half hi = l>>4,
//   element e -> K = hi*8 + (e&7) + (e>>3)*16  (per K-chunk of 32)
//   enc/dec packed: [32 Ntiles][6 Kchunks][32 lanes][16 halfs]
//   out    packed: [ 2 Ntiles][4 Kchunks][32 lanes][16 halfs]
// ---------------------------------------------------------------------------
__global__ void pack_kernel(
    const float* __restrict__ enc_Wih, const float* __restrict__ enc_Whh,
    const float* __restrict__ enc_bih, const float* __restrict__ enc_bhh,
    const float* __restrict__ dec_Wih, const float* __restrict__ dec_Whh,
    const float* __restrict__ dec_bih, const float* __restrict__ dec_bhh,
    const float* __restrict__ act_W,   const float* __restrict__ act_b,
    const float* __restrict__ time_W,  const float* __restrict__ time_b,
    _Float16* __restrict__ wenc, _Float16* __restrict__ wdec,
    _Float16* __restrict__ wout,
    float* __restrict__ benc, float* __restrict__ bdec, float* __restrict__ bout)
{
    const int NW = 32 * 6 * 32 * 16;   // 98304 elements per LSTM weight pack
    int idx = blockIdx.x * blockDim.x + threadIdx.x;

    if (idx < 2 * NW) {
        const float* Wih = (idx < NW) ? enc_Wih : dec_Wih;
        const float* Whh = (idx < NW) ? enc_Whh : dec_Whh;
        _Float16*    dst = (idx < NW) ? wenc : wdec;
        int i = (idx < NW) ? idx : idx - NW;
        int e = i & 15;
        int r = i >> 4;
        int lane = r & 31; r >>= 5;
        int kc = r % 6;
        int nt = r / 6;
        int n = nt * 16 + (lane & 15);
        int k = kc * 32 + (lane >> 4) * 8 + (e & 7) + ((e >> 3) << 4);
        float v = (k < E_) ? Wih[n * E_ + k] : Whh[n * H_ + (k - E_)];
        dst[i] = (_Float16)v;
    } else if (idx < 2 * NW + 4096) {
        int i = idx - 2 * NW;
        int e = i & 15;
        int r = i >> 4;
        int lane = r & 31; r >>= 5;
        int kc = r & 3;
        int nt = r >> 2;
        int n = nt * 16 + (lane & 15);
        int k = kc * 32 + (lane >> 4) * 8 + (e & 7) + ((e >> 3) << 4);
        float v = (n < C_) ? act_W[n * H_ + k] : time_W[k];
        wout[i] = (_Float16)v;
    } else if (idx < 2 * NW + 4096 + G_) {
        int n = idx - (2 * NW + 4096);
        benc[n] = enc_bih[n] + enc_bhh[n];
    } else if (idx < 2 * NW + 4096 + 2 * G_) {
        int n = idx - (2 * NW + 4096 + G_);
        bdec[n] = dec_bih[n] + dec_bhh[n];
    } else if (idx < 2 * NW + 4096 + 2 * G_ + 32) {
        int n = idx - (2 * NW + 4096 + 2 * G_);
        bout[n] = (n < C_) ? act_b[n] : time_b[0];
    }
}

// ---------------------------------------------------------------------------
// Kernel 2: x_emb = LN(leaky(x @ init_W.T + init_b)) -> f16   [B*S rows]
// ---------------------------------------------------------------------------
__global__ __launch_bounds__(256)
void embed_kernel(const float* __restrict__ x,
                  const float* __restrict__ W,   // [64,32]
                  const float* __restrict__ b,
                  const float* __restrict__ g,
                  const float* __restrict__ beta,
                  _Float16* __restrict__ xemb)
{
    __shared__ float sW[E_ * NF_];
    __shared__ float sb[E_], sg[E_], sbt[E_];
    for (int i = threadIdx.x; i < E_ * NF_; i += 256) sW[i] = W[i];
    if (threadIdx.x < E_) {
        sb[threadIdx.x]  = b[threadIdx.x];
        sg[threadIdx.x]  = g[threadIdx.x];
        sbt[threadIdx.x] = beta[threadIdx.x];
    }
    __syncthreads();

    size_t row = (size_t)blockIdx.x * 256 + threadIdx.x;   // < 262144
    const float* xr = x + row * NF_;
    float xv[NF_];
#pragma unroll
    for (int k = 0; k < NF_; ++k) xv[k] = xr[k];

    float y[E_];
    float mean = 0.0f;
#pragma unroll 4
    for (int e = 0; e < E_; ++e) {
        float s = sb[e];
        const float* wr = &sW[e * NF_];
#pragma unroll
        for (int k = 0; k < NF_; ++k) s += xv[k] * wr[k];
        s = (s >= 0.0f) ? s : 0.01f * s;
        y[e] = s;
        mean += s;
    }
    mean *= (1.0f / E_);
    float var = 0.0f;
#pragma unroll 4
    for (int e = 0; e < E_; ++e) { float d = y[e] - mean; var += d * d; }
    var *= (1.0f / E_);
    float rstd = rsqrtf(var + 1e-5f);

    _Float16* o = xemb + row * E_;
#pragma unroll 4
    for (int e = 0; e < E_; ++e)
        o[e] = (_Float16)((y[e] - mean) * rstd * sg[e] + sbt[e]);
}

// ---------------------------------------------------------------------------
// Kernel 3: batch-parallel LSTM.  16 workgroups x 16 batch rows.
//   gates[16,512] = [x_t | h] @ Wcat^T  via WMMA f16 (K=192 = 6 chunks)
//   - x-part of A (kc 0,1) loaded straight from global (L1-broadcast),
//     consumed LAST in the K loop so load latency hides under the h WMMAs
//   - c state lives in registers (same thread owns same 4 cells every step)
//   - accumulators initialized with the gate bias (per-lane constant column)
//   512 threads = 16 waves; wave w owns gate tiles {hl*16+jg, hl*16+8+jg}
//   mode 0 = encoder (h0=1,c0=0, write hT at end)
//   mode 1 = decoder (h0/c0 from global, write per-step h -> hs f16)
// ---------------------------------------------------------------------------
__global__ __launch_bounds__(512)
void lstm_kernel(const _Float16* __restrict__ xemb,   // [B,S,64] f16
                 const _Float16* __restrict__ wpk,    // packed [32][6][32][16]
                 const float*    __restrict__ bias,   // [512] bih+bhh
                 const float*    __restrict__ h0,     // [B,128] (mode 1)
                 const float*    __restrict__ c0,     // [B,128] (mode 1)
                 float*          __restrict__ hT,     // [B,128] (mode 0)
                 _Float16*       __restrict__ hs,     // [B,1023,128] (mode 1)
                 int steps, int mode)
{
    __shared__ __align__(16) _Float16 harow[16 * H_];   // h[m][j]
    __shared__ __align__(16) float    gates[16 * G_];   // [m][n] row-major

    const int b0   = blockIdx.x * 16;
    const int t    = threadIdx.x;
    const int lane = t & 31;
    const int w    = t >> 5;
    const int jg   = w & 7;
    const int hl   = w >> 3;
    const int nt0  = hl * 16 + jg;
    const int nt1  = hl * 16 + 8 + jg;
    const int mA   = lane & 15;     // A-row / D-column index held by this lane
    const int hiA  = lane >> 4;
    const float bias0 = bias[nt0 * 16 + mA];
    const float bias1 = bias[nt1 * 16 + mA];

    // init h (LDS) and c (registers)
    for (int i = t; i < 16 * H_; i += 512)
        harow[i] = (mode == 0) ? (_Float16)1.0f
                               : (_Float16)h0[(b0 + (i >> 7)) * H_ + (i & 127)];
    float creg[4];
#pragma unroll
    for (int q = 0; q < 4; ++q) {
        int idx = t + q * 512;
        creg[q] = (mode == 0) ? 0.0f : c0[(b0 + (idx >> 7)) * H_ + (idx & 127)];
    }

    const _Float16* xrow = xemb + (size_t)(b0 + mA) * S_ * E_;

    __syncthreads();

    for (int s = 0; s < steps; ++s) {
        // ---- A x-fragments (kc 0,1) straight from global ----
        const _Float16* xs = xrow + (size_t)s * E_;
        Frag16 ax0, ax1;
        ax0.q[0] = *(const uint4*)(xs + hiA * 8);
        ax0.q[1] = *(const uint4*)(xs + 16 + hiA * 8);
        ax1.q[0] = *(const uint4*)(xs + 32 + hiA * 8);
        ax1.q[1] = *(const uint4*)(xs + 48 + hiA * 8);
        if (s + 1 < steps) __builtin_prefetch(xs + E_, 0, 0);

        v8f acc0, acc1;
#pragma unroll
        for (int r = 0; r < 8; ++r) { acc0[r] = bias0; acc1[r] = bias1; }

        // K-chunk order: h-part (LDS, kc 2..5) first, x-part (global) last
#pragma unroll
        for (int kk = 0; kk < 6; ++kk) {
            const int kc = (kk < 4) ? (kk + 2) : (kk - 4);
            Frag16 a;
            if (kc == 0)      a = ax0;
            else if (kc == 1) a = ax1;
            else {
#pragma unroll
                for (int j = 0; j < 8; ++j) {
                    int k = (kc - 2) * 32 + a_koff(j, hiA);
                    a.u[j] = *(const uint32_t*)&harow[mA * H_ + k];
                }
            }
            Frag16 bf;
            const uint4* p0 = (const uint4*)(wpk + (((nt0 * 6 + kc) * 32 + lane) << 4));
            bf.q[0] = p0[0]; bf.q[1] = p0[1];
            acc0 = __builtin_amdgcn_wmma_f32_16x16x32_f16(
                       false, a.h, false, bf.h, (short)0, acc0, false, false);
            const uint4* p1 = (const uint4*)(wpk + (((nt1 * 6 + kc) * 32 + lane) << 4));
            bf.q[0] = p1[0]; bf.q[1] = p1[1];
            acc1 = __builtin_amdgcn_wmma_f32_16x16x32_f16(
                       false, a.h, false, bf.h, (short)0, acc1, false, false);
        }

        // ---- scatter gate tiles (D layout: lane holds col mA, rows r+hiA*8) ----
#pragma unroll
        for (int r = 0; r < 8; ++r) {
            int m = r + hiA * 8;
            gates[m * G_ + nt0 * 16 + mA] = acc0[r];
            gates[m * G_ + nt1 * 16 + mA] = acc1[r];
        }
        __syncthreads();

        // ---- elementwise LSTM cell update (f32, c in registers) ----
#pragma unroll
        for (int q = 0; q < 4; ++q) {
            int idx = t + q * 512;          // 0..2047
            int m = idx >> 7, j = idx & 127;
            float gi = gates[m * G_ + j];
            float gf = gates[m * G_ + H_ + j];
            float gg = gates[m * G_ + 2 * H_ + j];
            float go = gates[m * G_ + 3 * H_ + j];
            float c  = fsig(gf) * creg[q] + fsig(gi) * ftanh(gg);
            float h  = fsig(go) * ftanh(c);
            creg[q] = c;
            harow[m * H_ + j] = (_Float16)h;
            if (mode == 1)
                hs[((size_t)(b0 + m) * SD + s) * H_ + j] = (_Float16)h;
            else if (s == steps - 1)
                hT[(b0 + m) * H_ + j] = h;
        }
        __syncthreads();
    }
}

// ---------------------------------------------------------------------------
// Kernel 4: latent path (per batch row): mu/logvar (+LN), latent, dh0/dc0 (+LN)
// ---------------------------------------------------------------------------
__global__ __launch_bounds__(128)
void latent_kernel(const float* __restrict__ hT, const float* __restrict__ eps,
                   const float* __restrict__ mu_W, const float* __restrict__ mu_b,
                   const float* __restrict__ mu_g, const float* __restrict__ mu_beta,
                   const float* __restrict__ lv_W, const float* __restrict__ lv_b,
                   const float* __restrict__ lv_g, const float* __restrict__ lv_beta,
                   const float* __restrict__ lh_W, const float* __restrict__ lh_b,
                   const float* __restrict__ lh_g, const float* __restrict__ lh_beta,
                   const float* __restrict__ lc_W, const float* __restrict__ lc_b,
                   const float* __restrict__ lc_g, const float* __restrict__ lc_beta,
                   float* __restrict__ out, float* __restrict__ dh0,
                   float* __restrict__ dc0)
{
    __shared__ float sh[H_], pre[H_], stats[8], lat[L_], smu[L_], slv[L_];
    const int b = blockIdx.x, t = threadIdx.x;

    sh[t] = hT[b * H_ + t];
    __syncthreads();

    {   // mu_pre (threads 0-63) / lv_pre (threads 64-127)
        int j = t & 63;
        const float* W = (t < 64) ? mu_W : lv_W;
        float s        = (t < 64) ? mu_b[j] : lv_b[j];
        for (int k = 0; k < H_; ++k) s += sh[k] * W[j * H_ + k];
        pre[t] = (s >= 0.0f) ? s : 0.01f * s;
    }
    __syncthreads();
    if (t < 2) {
        float m = 0.0f;
        for (int k = 0; k < 64; ++k) m += pre[t * 64 + k];
        m *= (1.0f / 64.0f);
        float v = 0.0f;
        for (int k = 0; k < 64; ++k) { float d = pre[t * 64 + k] - m; v += d * d; }
        v *= (1.0f / 64.0f);
        stats[t * 2]     = m;
        stats[t * 2 + 1] = rsqrtf(v + 1e-5f);
    }
    __syncthreads();
    {
        int j = t & 63;
        if (t < 64) {
            float mu = (pre[t] - stats[0]) * stats[1] * mu_g[j] + mu_beta[j];
            smu[j] = mu;
            out[MU_OFF + b * L_ + j] = mu;
        } else {
            float lv = (pre[t] - stats[2]) * stats[3] * lv_g[j] + lv_beta[j];
            slv[j] = lv;
            out[LV_OFF + b * L_ + j] = lv;
        }
    }
    __syncthreads();
    if (t < 64) lat[t] = smu[t] + eps[b * L_ + t] * __expf(0.5f * slv[t]);
    __syncthreads();

    // dh0
    {
        float s = lh_b[t];
        for (int k = 0; k < L_; ++k) s += lat[k] * lh_W[t * L_ + k];
        pre[t] = (s >= 0.0f) ? s : 0.01f * s;
    }
    __syncthreads();
    if (t == 0) {
        float m = 0.0f;
        for (int k = 0; k < H_; ++k) m += pre[k];
        m *= (1.0f / H_);
        float v = 0.0f;
        for (int k = 0; k < H_; ++k) { float d = pre[k] - m; v += d * d; }
        v *= (1.0f / H_);
        stats[4] = m; stats[5] = rsqrtf(v + 1e-5f);
    }
    __syncthreads();
    dh0[b * H_ + t] = (pre[t] - stats[4]) * stats[5] * lh_g[t] + lh_beta[t];
    __syncthreads();

    // dc0
    {
        float s = lc_b[t];
        for (int k = 0; k < L_; ++k) s += lat[k] * lc_W[t * L_ + k];
        pre[t] = (s >= 0.0f) ? s : 0.01f * s;
    }
    __syncthreads();
    if (t == 0) {
        float m = 0.0f;
        for (int k = 0; k < H_; ++k) m += pre[k];
        m *= (1.0f / H_);
        float v = 0.0f;
        for (int k = 0; k < H_; ++k) { float d = pre[k] - m; v += d * d; }
        v *= (1.0f / H_);
        stats[6] = m; stats[7] = rsqrtf(v + 1e-5f);
    }
    __syncthreads();
    dc0[b * H_ + t] = (pre[t] - stats[6]) * stats[7] * lc_g[t] + lc_beta[t];
}

// ---------------------------------------------------------------------------
// Kernel 5: [acts | ts] = hs @ [act_W ; time_W]^T + bias   via WMMA
//   rows = B*(S-1) = 261888 = 16368 M-tiles; N = 32 (2 tiles); K = 128 (4 chunks)
// ---------------------------------------------------------------------------
__global__ __launch_bounds__(256)
void out_kernel(const _Float16* __restrict__ hs,    // [261888,128] f16
                const _Float16* __restrict__ wpk,   // packed [2][4][32][16]
                const float*    __restrict__ bias,  // [32]
                float* __restrict__ out)
{
    const int lane = threadIdx.x & 31;
    const int w    = threadIdx.x >> 5;
    const int mt   = blockIdx.x * 8 + w;     // M-tile 0..16367
    const int mA   = lane & 15;
    const int hiA  = lane >> 4;
    const size_t rowA = (size_t)mt * 16 + mA;

    v8f acc0 = {};
    v8f acc1 = {};
#pragma unroll
    for (int kc = 0; kc < 4; ++kc) {
        Frag16 a;
#pragma unroll
        for (int j = 0; j < 8; ++j) {
            int k = kc * 32 + a_koff(j, hiA);
            a.u[j] = *(const uint32_t*)&hs[rowA * H_ + k];
        }
        Frag16 bf;
        const uint4* p0 = (const uint4*)(wpk + ((kc * 32 + lane) << 4));
        bf.q[0] = p0[0]; bf.q[1] = p0[1];
        acc0 = __builtin_amdgcn_wmma_f32_16x16x32_f16(
                   false, a.h, false, bf.h, (short)0, acc0, false, false);
        const uint4* p1 = (const uint4*)(wpk + (((4 + kc) * 32 + lane) << 4));
        bf.q[0] = p1[0]; bf.q[1] = p1[1];
        acc1 = __builtin_amdgcn_wmma_f32_16x16x32_f16(
                   false, a.h, false, bf.h, (short)0, acc1, false, false);
    }

    const float bb0 = bias[mA];        // n in [0,16)  -> acts cols 0..15
    const float bb1 = bias[16 + mA];   // n in [16,32) -> acts 16..30 or ts (31)
#pragma unroll
    for (int r = 0; r < 8; ++r) {
        int m = r + hiA * 8;
        size_t row = (size_t)mt * 16 + m;
        out[row * C_ + mA] = acc0[r] + bb0;
        int n1 = 16 + mA;
        float v1 = acc1[r] + bb1;
        if (n1 < C_) out[row * C_ + n1] = v1;
        else         out[TS_OFF + row]  = v1;
    }
}

// ---------------------------------------------------------------------------
// Host-side launcher
// ---------------------------------------------------------------------------
extern "C" void kernel_launch(void* const* d_in, const int* in_sizes, int n_in,
                              void* d_out, int out_size, void* d_ws, size_t ws_size,
                              hipStream_t stream) {
    const float* x        = (const float*)d_in[0];
    const float* eps      = (const float*)d_in[1];
    const float* init_W   = (const float*)d_in[2];
    const float* init_b   = (const float*)d_in[3];
    const float* init_g   = (const float*)d_in[4];
    const float* init_bt  = (const float*)d_in[5];
    const float* enc_Wih  = (const float*)d_in[6];
    const float* enc_Whh  = (const float*)d_in[7];
    const float* enc_bih  = (const float*)d_in[8];
    const float* enc_bhh  = (const float*)d_in[9];
    const float* mu_W     = (const float*)d_in[10];
    const float* mu_b     = (const float*)d_in[11];
    const float* mu_g     = (const float*)d_in[12];
    const float* mu_beta  = (const float*)d_in[13];
    const float* lv_W     = (const float*)d_in[14];
    const float* lv_b     = (const float*)d_in[15];
    const float* lv_g     = (const float*)d_in[16];
    const float* lv_beta  = (const float*)d_in[17];
    const float* dec_Wih  = (const float*)d_in[18];
    const float* dec_Whh  = (const float*)d_in[19];
    const float* dec_bih  = (const float*)d_in[20];
    const float* dec_bhh  = (const float*)d_in[21];
    const float* lh_W     = (const float*)d_in[22];
    const float* lh_b     = (const float*)d_in[23];
    const float* lh_g     = (const float*)d_in[24];
    const float* lh_beta  = (const float*)d_in[25];
    const float* lc_W     = (const float*)d_in[26];
    const float* lc_b     = (const float*)d_in[27];
    const float* lc_g     = (const float*)d_in[28];
    const float* lc_beta  = (const float*)d_in[29];
    const float* act_W    = (const float*)d_in[30];
    const float* act_b    = (const float*)d_in[31];
    const float* time_W   = (const float*)d_in[32];
    const float* time_b   = (const float*)d_in[33];
    float* out = (float*)d_out;

    // Workspace carve-up (256B aligned)
    char* ws = (char*)d_ws;
    size_t off = 0;
    auto carve = [&](size_t bytes) {
        char* p = ws + off;
        off += (bytes + 255) & ~(size_t)255;
        return p;
    };
    _Float16* xemb = (_Float16*)carve((size_t)B_ * S_ * E_ * 2);      // 33.6 MB
    _Float16* hs   = (_Float16*)carve((size_t)B_ * SD * H_ * 2);      // 67.0 MB
    _Float16* wenc = (_Float16*)carve(32 * 6 * 32 * 16 * 2);          // 192 KB
    _Float16* wdec = (_Float16*)carve(32 * 6 * 32 * 16 * 2);          // 192 KB
    _Float16* wout = (_Float16*)carve(2 * 4 * 32 * 16 * 2);           // 8 KB
    float*    benc = (float*)carve(G_ * 4);
    float*    bdec = (float*)carve(G_ * 4);
    float*    bout = (float*)carve(32 * 4);
    float*    hT   = (float*)carve((size_t)B_ * H_ * 4);
    float*    dh0  = (float*)carve((size_t)B_ * H_ * 4);
    float*    dc0  = (float*)carve((size_t)B_ * H_ * 4);
    (void)ws_size; (void)in_sizes; (void)n_in; (void)out_size;

    // 1) pack weights / biases
    {
        int total = 2 * (32 * 6 * 32 * 16) + 4096 + 2 * G_ + 32;
        int blocks = (total + 255) / 256;
        pack_kernel<<<blocks, 256, 0, stream>>>(
            enc_Wih, enc_Whh, enc_bih, enc_bhh,
            dec_Wih, dec_Whh, dec_bih, dec_bhh,
            act_W, act_b, time_W, time_b,
            wenc, wdec, wout, benc, bdec, bout);
    }

    // 2) embed:  x -> x_emb (f16)
    embed_kernel<<<(B_ * S_) / 256, 256, 0, stream>>>(
        x, init_W, init_b, init_g, init_bt, xemb);

    // 3) encoder LSTM (h0=1, c0=0) -> hT
    lstm_kernel<<<B_ / 16, 512, 0, stream>>>(
        xemb, wenc, benc, nullptr, nullptr, hT, nullptr, S_, 0);

    // 4) latent path: mu/logvar -> out; latent -> dh0, dc0
    latent_kernel<<<B_, H_, 0, stream>>>(
        hT, eps,
        mu_W, mu_b, mu_g, mu_beta,
        lv_W, lv_b, lv_g, lv_beta,
        lh_W, lh_b, lh_g, lh_beta,
        lc_W, lc_b, lc_g, lc_beta,
        out, dh0, dc0);

    // 5) decoder LSTM -> hs (f16)
    lstm_kernel<<<B_ / 16, 512, 0, stream>>>(
        xemb, wdec, bdec, dh0, dc0, nullptr, hs, SD, 1);

    // 6) output GEMM: hs -> acts + ts
    out_kernel<<<(B_ * SD / 16) / 8, 256, 0, stream>>>(hs, wout, bout, out);
}
